// RelativePositionAttention_8134668058977
// MI455X (gfx1250) — matrix operand order repaired and checked by reading, not compiled, not run on hardware
//
#include <hip/hip_runtime.h>
#include <cstdint>

// ---------------- problem constants ----------------
static constexpr int BB    = 4;     // batch
static constexpr int SEQ_N = 1024;  // sequence
static constexpr int DM    = 512;   // d_model
static constexpr int NH    = 8;     // heads
static constexpr int DH    = 64;    // head dim
static constexpr int MROWS = BB * SEQ_N; // 4096

typedef __attribute__((ext_vector_type(16))) __bf16 v16bf;
typedef __attribute__((ext_vector_type(8)))  __bf16 v8bf;
typedef __attribute__((ext_vector_type(8)))  float  v8f;

// ---------------- WMMA / cross-lane helpers ----------------
__device__ __forceinline__ v8f wmma_bf16(v16bf a, v16bf b, v8f c) {
  return __builtin_amdgcn_wmma_f32_16x16x32_bf16(false, a, false, b, (short)0, c, false, false);
}

// swap values between lanes L and L^16 (the two 16-lane rows of a wave32)
__device__ __forceinline__ float xor16f(float x) {
#if __has_builtin(__builtin_amdgcn_permlanex16)
  int i = __builtin_bit_cast(int, x);
  i = __builtin_amdgcn_permlanex16(i, i, 0x76543210, 0xfedcba98, false, false);
  return __builtin_bit_cast(float, i);
#else
  return __shfl_xor(x, 16, 32);
#endif
}

// A-matrix fragment (16x32 bf16), source row-major [M][lda].
// Lane L<16: row m0+L, K = kb+{0..15}; lane L>=16: row m0+(L-16), K = kb+{16..31}.
__device__ __forceinline__ v16bf load_fragA(const __bf16* A, int lda, int m0, int kb, int lane) {
  int lr = lane & 15, hi = lane >> 4;
  const __bf16* p = A + (size_t)(m0 + lr) * lda + kb + hi * 16;
  v8bf a = *(const v8bf*)p;
  v8bf b = *(const v8bf*)(p + 8);
  v16bf r;
#pragma unroll
  for (int i = 0; i < 8; ++i) { r[i] = a[i]; r[i + 8] = b[i]; }
  return r;
}

// B-matrix fragment (32x16 bf16) from a TRANSPOSED operand Wt[N][ldb] (ldb = K stride).
// Bmat[k][n] = Wt[n0+n][kb+k]. Lane L: n = L&15, k = (L>>4)*16 + {0..15} (contiguous).
__device__ __forceinline__ v16bf load_fragB_T(const __bf16* Wt, int ldb, int n0, int kb, int lane) {
  int lr = lane & 15, hi = lane >> 4;
  const __bf16* p = Wt + (size_t)(n0 + lr) * ldb + kb + hi * 16;
  v8bf a = *(const v8bf*)p;
  v8bf b = *(const v8bf*)(p + 8);
  v16bf r;
#pragma unroll
  for (int i = 0; i < 8; ++i) { r[i] = a[i]; r[i + 8] = b[i]; }
  return r;
}

// ---------------- QR per head: P = Uq * Uq^T ----------------
__global__ void __launch_bounds__(64) qr_kernel(const float* __restrict__ U,
                                                float* __restrict__ P) {
  __shared__ float A[64][64];
  __shared__ float s_scale;
  int h = blockIdx.x, t = threadIdx.x; // 64 threads: t = row
  const float* Uh = U + (size_t)h * 64 * 64;
  for (int c = 0; c < 64; ++c) A[t][c] = Uh[t * 64 + c];
  __syncthreads();
  for (int j = 0; j < 64; ++j) {
    if (t == 0) {
      float s = 0.f;
      for (int r = 0; r < 64; ++r) s += A[r][j] * A[r][j];
      s_scale = rsqrtf(fmaxf(s, 1e-30f));
    }
    __syncthreads();
    A[t][j] *= s_scale;
    __syncthreads();
    if (t > j) {               // thread t owns column t
      float dot = 0.f;
      for (int r = 0; r < 64; ++r) dot += A[r][j] * A[r][t];
      for (int r = 0; r < 64; ++r) A[r][t] -= dot * A[r][j];
    }
    __syncthreads();
  }
  float* Ph = P + (size_t)h * 64 * 64;
  for (int f = 0; f < 64; ++f) {
    float s = 0.f;
    for (int e = 0; e < 64; ++e) s += A[t][e] * A[f][e];
    Ph[t * 64 + f] = s;
  }
}

// ---------------- fold P into projection weight, output transposed bf16 ----------------
__global__ void __launch_bounds__(256) fold_kernel(const float* __restrict__ W,
                                                   const float* __restrict__ P,
                                                   __bf16* __restrict__ Wt) {
  int idx = blockIdx.x * blockDim.x + threadIdx.x;
  if (idx >= DM * DM) return;
  int c = idx / DM, d = idx % DM;
  int h = c >> 6, f = c & 63;
  const float* Ph = P + (size_t)h * 64 * 64;
  const float* Wr = W + (size_t)d * DM + h * 64;
  float s = 0.f;
#pragma unroll 8
  for (int e = 0; e < 64; ++e) s += Wr[e] * Ph[e * 64 + f];
  Wt[(size_t)c * DM + d] = (__bf16)s;
}

__global__ void __launch_bounds__(256) tconv_kernel(const float* __restrict__ W,
                                                    __bf16* __restrict__ Wt,
                                                    int rows, int cols) {
  int idx = blockIdx.x * blockDim.x + threadIdx.x;
  if (idx >= rows * cols) return;
  int c = idx / rows, r = idx % rows;
  Wt[(size_t)c * rows + r] = (__bf16)W[(size_t)r * cols + c];
}

__global__ void __launch_bounds__(256) f2bf_kernel(const float* __restrict__ in,
                                                   __bf16* __restrict__ out, int n) {
  int idx = blockIdx.x * blockDim.x + threadIdx.x;
  if (idx < n) out[idx] = (__bf16)in[idx];
}

// ---------------- WMMA GEMM: C[MxN] = A[MxK] * Bt[NxK]^T, 16x32 per wave ----------------
// MODE 0: bf16 C row-major   MODE 1: f32 C row-major   MODE 2: bf16 V-transposed [b,h,d,n]
template <int MODE>
__global__ void __launch_bounds__(128)
gemm_kernel(const __bf16* __restrict__ A, const __bf16* __restrict__ Bt,
            void* __restrict__ C, int M, int K, int Ncols) {
  int lane = threadIdx.x & 31, wave = threadIdx.x >> 5;
  int n0 = blockIdx.x * 32;
  int m0 = (blockIdx.y * 4 + wave) * 16;
  if (m0 >= M || n0 >= Ncols) return;
  v8f acc[2] = {};
  for (int kb = 0; kb < K; kb += 32) {
    v16bf a = load_fragA(A, K, m0, kb, lane);
    acc[0] = wmma_bf16(a, load_fragB_T(Bt, K, n0,      kb, lane), acc[0]);
    acc[1] = wmma_bf16(a, load_fragB_T(Bt, K, n0 + 16, kb, lane), acc[1]);
  }
  int lr = lane & 15, hi = lane >> 4;
#pragma unroll
  for (int nn = 0; nn < 2; ++nn) {
    int nc = n0 + nn * 16 + lr;
    if (MODE == 0) {
      __bf16* Cb = (__bf16*)C;
#pragma unroll
      for (int r = 0; r < 8; ++r)
        Cb[(size_t)(m0 + r + 8 * hi) * Ncols + nc] = (__bf16)acc[nn][r];
    } else if (MODE == 1) {
      float* Cf = (float*)C;
#pragma unroll
      for (int r = 0; r < 8; ++r)
        Cf[(size_t)(m0 + r + 8 * hi) * Ncols + nc] = acc[nn][r];
    } else {
      __bf16* Cb = (__bf16*)C;
#pragma unroll
      for (int r = 0; r < 8; ++r) {
        int m = m0 + r + 8 * hi;
        int b = m / SEQ_N, n = m % SEQ_N;
        int h = nc >> 6, d = nc & 63;
        Cb[(((size_t)b * NH + h) * DH + d) * SEQ_N + n] = (__bf16)acc[nn][r];
      }
    }
  }
}

// ---------------- RoPE tables ----------------
__global__ void __launch_bounds__(256) angles_kernel(const float* __restrict__ pos,
                                                     const float* __restrict__ freqs,
                                                     float* __restrict__ cosT,
                                                     float* __restrict__ sinT) {
  int idx = blockIdx.x * blockDim.x + threadIdx.x; // H*N*32
  if (idx >= NH * SEQ_N * 32) return;
  int h = idx / (SEQ_N * 32);
  int rem = idx % (SEQ_N * 32);
  int n = rem / 32, m = rem % 32;
  float a = pos[n * 2 + 0] * freqs[(h * 32 + m) * 2 + 0]
          + pos[n * 2 + 1] * freqs[(h * 32 + m) * 2 + 1];
  cosT[idx] = __cosf(a);
  sinT[idx] = __sinf(a);
}

// rotate [b,n,h,d] bf16 -> [b,h,n,d] bf16, output multiplied by `scale`
__global__ void __launch_bounds__(256) rope_kernel(const __bf16* __restrict__ X,
                                                   const float* __restrict__ cosT,
                                                   const float* __restrict__ sinT,
                                                   __bf16* __restrict__ Xr, float scale) {
  int idx = blockIdx.x * blockDim.x + threadIdx.x; // B*H*N*32
  if (idx >= BB * NH * SEQ_N * 32) return;
  int b = idx / (NH * SEQ_N * 32);
  int rem = idx % (NH * SEQ_N * 32);
  int h = rem / (SEQ_N * 32);
  rem = rem % (SEQ_N * 32);
  int n = rem / 32, m = rem % 32;
  size_t src = ((size_t)(b * SEQ_N + n) * NH + h) * DH + 2 * m;
  float e = (float)X[src], o = (float)X[src + 1];
  float c = cosT[(h * SEQ_N + n) * 32 + m];
  float s = sinT[(h * SEQ_N + n) * 32 + m];
  size_t dst = (((size_t)b * NH + h) * SEQ_N + n) * DH + 2 * m;
  Xr[dst]     = (__bf16)((e * c - o * s) * scale);
  Xr[dst + 1] = (__bf16)((e * s + o * c) * scale);
}

// ---------------- fused flash attention (transposed-score form) ----------------
// grid = (B*H, N/128), block = 128 (4 waves). Wave -> 32 queries, 32 keys / iter.
// S' = K Q^T : D-layout => lane owns one query column; keys live in v8f elements.
// O' = V^T P^T : per-query softmax state is a scalar per lane. No LDS, no barriers.
__global__ void __launch_bounds__(128)
__attribute__((amdgpu_waves_per_eu(1)))
flash_kernel(const __bf16* __restrict__ Qr, const __bf16* __restrict__ Kr,
             const __bf16* __restrict__ Vt, __bf16* __restrict__ Oattn) {
  int lane = threadIdx.x & 31, wave = threadIdx.x >> 5;
  int lr = lane & 15, hi = lane >> 4;
  int bh = blockIdx.x;
  int b = bh >> 3, h = bh & 7;
  int m0 = (blockIdx.y * 4 + wave) * 32; // first query of this wave

  const __bf16* Qb = Qr + (size_t)bh * SEQ_N * DH; // [n][d] (pre-scaled by 1/sqrt(32))
  const __bf16* Kb = Kr + (size_t)bh * SEQ_N * DH; // [n][d]
  const __bf16* Vb = Vt + (size_t)bh * DH * SEQ_N; // [d][n]

  // Q as B-fragments (d x queries), loaded once: [query tile][d slice]
  v16bf qb[2][2];
#pragma unroll
  for (int qt = 0; qt < 2; ++qt)
#pragma unroll
    for (int dsl = 0; dsl < 2; ++dsl)
      qb[qt][dsl] = load_fragB_T(Qb, DH, m0 + qt * 16, dsl * 32, lane);

  v8f o[2][4] = {};            // O' accum: [query tile][d tile], lane = query column
  float m_run[2] = {-1e30f, -1e30f};
  float l_run[2] = {0.f, 0.f};

  for (int jt = 0; jt < SEQ_N; jt += 32) {
    if (jt + 32 < SEQ_N) {
      __builtin_prefetch(Kb + (size_t)(jt + 32 + lane) * DH, 0, 3);
      __builtin_prefetch(Vb + (size_t)lane * SEQ_N + jt + 32, 0, 3);
      __builtin_prefetch(Vb + (size_t)(lane + 32) * SEQ_N + jt + 32, 0, 3);
    }
    // K as A-fragments: [key tile][d slice]
    v16bf ka[2][2];
#pragma unroll
    for (int kt = 0; kt < 2; ++kt)
#pragma unroll
      for (int dsl = 0; dsl < 2; ++dsl)
        ka[kt][dsl] = load_fragA(Kb, DH, jt + kt * 16, dsl * 32, lane);
    // V^T as A-fragments: [d tile] (keys jt..jt+31 in elements)
    v16bf va[4];
#pragma unroll
    for (int t = 0; t < 4; ++t)
      va[t] = load_fragA(Vb, SEQ_N, t * 16, jt, lane);

#pragma unroll
    for (int qt = 0; qt < 2; ++qt) {
      // scores transposed: s0 = keys jt+{r+8*hi}, s1 = keys jt+16+{r+8*hi}
      v8f s0 = {}, s1 = {};
      s0 = wmma_bf16(ka[0][0], qb[qt][0], s0);
      s0 = wmma_bf16(ka[0][1], qb[qt][1], s0);
      s1 = wmma_bf16(ka[1][0], qb[qt][0], s1);
      s1 = wmma_bf16(ka[1][1], qb[qt][1], s1);

      // per-query (per-lane) online softmax
      float mx = s0[0];
#pragma unroll
      for (int r = 1; r < 8; ++r) mx = fmaxf(mx, s0[r]);
#pragma unroll
      for (int r = 0; r < 8; ++r) mx = fmaxf(mx, s1[r]);
      mx = fmaxf(mx, xor16f(mx));                 // combine the two key halves
      float mnew = fmaxf(m_run[qt], mx);
      float alpha = __expf(m_run[qt] - mnew);
      m_run[qt] = mnew;

      float p0[8], p1[8], sum = 0.f;
#pragma unroll
      for (int r = 0; r < 8; ++r) { p0[r] = __expf(s0[r] - mnew); sum += p0[r]; }
#pragma unroll
      for (int r = 0; r < 8; ++r) { p1[r] = __expf(s1[r] - mnew); sum += p1[r]; }
      sum += xor16f(sum);
      l_run[qt] = l_run[qt] * alpha + sum;

#pragma unroll
      for (int t = 0; t < 4; ++t)
#pragma unroll
        for (int r = 0; r < 8; ++r) o[qt][t][r] *= alpha;

      // assemble P^T B-fragment: lane needs keys hi*16+{0..15} for its query.
      // hi=0: elems i = own p0 (keys i), elems i+8 = partner p0 (keys 8+i)
      // hi=1: elems i = partner p1 (keys 16+i), elems i+8 = own p1 (keys 24+i)
      v16bf pb;
#pragma unroll
      for (int r = 0; r < 8; ++r) {
        float x0 = xor16f(p0[r]);
        float x1 = xor16f(p1[r]);
        pb[r]     = (__bf16)(hi ? x1 : p0[r]);
        pb[r + 8] = (__bf16)(hi ? p1[r] : x0);
      }

      // O' += V^T @ P^T over 4 d-tiles
#pragma unroll
      for (int t = 0; t < 4; ++t)
        o[qt][t] = wmma_bf16(va[t], pb, o[qt][t]);
    }
  }

  // epilogue: per-lane 1/l, store [b, n=query, h*64+d] bf16
#pragma unroll
  for (int qt = 0; qt < 2; ++qt) {
    float inv = 1.f / l_run[qt];
    int q = m0 + qt * 16 + lr;
#pragma unroll
    for (int t = 0; t < 4; ++t)
#pragma unroll
      for (int r = 0; r < 8; ++r) {
        int d = t * 16 + r + 8 * hi;
        Oattn[((size_t)b * SEQ_N + q) * DM + h * DH + d] = (__bf16)(o[qt][t][r] * inv);
      }
  }
}

// ---------------- host launch ----------------
extern "C" void kernel_launch(void* const* d_in, const int* in_sizes, int n_in,
                              void* d_out, int out_size, void* d_ws, size_t ws_size,
                              hipStream_t stream) {
  const float* x     = (const float*)d_in[0];
  const float* pos   = (const float*)d_in[1];
  const float* W_Q   = (const float*)d_in[2];
  const float* W_K   = (const float*)d_in[3];
  const float* W_V   = (const float*)d_in[4];
  const float* W_O   = (const float*)d_in[5];
  const float* U     = (const float*)d_in[6];
  const float* freqs = (const float*)d_in[7];
  float* out = (float*)d_out;

  char* w = (char*)d_ws;
  size_t off = 0;
  auto take = [&](size_t bytes) { void* p = w + off; off = (off + bytes + 255) & ~size_t(255); return p; };
  float*  P    = (float*) take((size_t)NH * 64 * 64 * 4);
  __bf16* WtQ  = (__bf16*)take((size_t)DM * DM * 2);
  __bf16* WtK  = (__bf16*)take((size_t)DM * DM * 2);
  __bf16* WtV  = (__bf16*)take((size_t)DM * DM * 2);
  __bf16* WtO  = (__bf16*)take((size_t)DM * DM * 2);
  __bf16* xbf  = (__bf16*)take((size_t)MROWS * DM * 2);
  __bf16* Qbf  = (__bf16*)take((size_t)MROWS * DM * 2);
  __bf16* Kbf  = (__bf16*)take((size_t)MROWS * DM * 2);
  __bf16* Vt   = (__bf16*)take((size_t)MROWS * DM * 2);
  float*  cosT = (float*) take((size_t)NH * SEQ_N * 32 * 4);
  float*  sinT = (float*) take((size_t)NH * SEQ_N * 32 * 4);
  __bf16* Qr   = (__bf16*)take((size_t)MROWS * DM * 2);
  __bf16* Kr   = (__bf16*)take((size_t)MROWS * DM * 2);
  __bf16* attn = (__bf16*)take((size_t)MROWS * DM * 2);
  (void)ws_size; (void)in_sizes; (void)n_in; (void)out_size;

  qr_kernel<<<NH, 64, 0, stream>>>(U, P);

  {
    int n = DM * DM, blk = 256, g = (n + blk - 1) / blk;
    fold_kernel<<<g, blk, 0, stream>>>(W_Q, P, WtQ);
    fold_kernel<<<g, blk, 0, stream>>>(W_K, P, WtK);
    tconv_kernel<<<g, blk, 0, stream>>>(W_V, WtV, DM, DM);
    tconv_kernel<<<g, blk, 0, stream>>>(W_O, WtO, DM, DM);
  }
  {
    int n = MROWS * DM, blk = 256, g = (n + blk - 1) / blk;
    f2bf_kernel<<<g, blk, 0, stream>>>(x, xbf, n);
  }

  dim3 gg(DM / 32, MROWS / 64), gb(128);
  gemm_kernel<0><<<gg, gb, 0, stream>>>(xbf, WtQ, Qbf, MROWS, DM, DM);
  gemm_kernel<0><<<gg, gb, 0, stream>>>(xbf, WtK, Kbf, MROWS, DM, DM);
  gemm_kernel<2><<<gg, gb, 0, stream>>>(xbf, WtV, Vt,  MROWS, DM, DM);

  {
    int n = NH * SEQ_N * 32, blk = 256, g = (n + blk - 1) / blk;
    angles_kernel<<<g, blk, 0, stream>>>(pos, freqs, cosT, sinT);
  }
  {
    int n = BB * NH * SEQ_N * 32, blk = 256, g = (n + blk - 1) / blk;
    const float sc = 0.17677669529663687f; // (dh/2)^-0.5
    rope_kernel<<<g, blk, 0, stream>>>(Qbf, cosT, sinT, Qr, sc);
    rope_kernel<<<g, blk, 0, stream>>>(Kbf, cosT, sinT, Kr, 1.0f);
  }

  flash_kernel<<<dim3(BB * NH, SEQ_N / 128), 128, 0, stream>>>(Qr, Kr, Vt, attn);

  gemm_kernel<1><<<gg, gb, 0, stream>>>(attn, WtO, out, MROWS, DM, DM);
}
